// CenterRegressor_14731737825835
// MI455X (gfx1250) — matrix operand order repaired and checked by reading, not compiled, not run on hardware
//
#include <hip/hip_runtime.h>
#include <hip/hip_bf16.h>

#define N_NODES 50000
#define N_EDGES 800000
#define IN_DIM  16
#define HID     256
#define N_LAYERS 4
#define ROW_TILES (N_NODES / 16)   // 3125 (exact)
#define KT 8                       // 256 / 32 k-tiles
#define NT 16                      // 256 / 16 n-tiles

typedef __attribute__((ext_vector_type(16))) __bf16 v16bf;
typedef __attribute__((ext_vector_type(8)))  float  v8f;

typedef int v4i __attribute__((vector_size(16)));   // matches builtin V4i

struct alignas(16) U4 { unsigned int x, y, z, w; };
union BF16x16 { v16bf v; U4 q[2]; };   // 32 bytes = 8 VGPRs of packed bf16

__device__ __forceinline__ unsigned short f2bf(float f) {
    unsigned int u = __float_as_uint(f);
    u += 0x7fffu + ((u >> 16) & 1u);           // round-to-nearest-even
    return (unsigned short)(u >> 16);
}

// ---------------------------------------------------------------- utilities
__global__ void k_zero(float* __restrict__ p, int n) {
    int i = blockIdx.x * blockDim.x + threadIdx.x;
    if (i < n) p[i] = 0.0f;
}

__global__ void k_degree(const long long* __restrict__ dst, float* __restrict__ deg, int ne) {
    int e = blockIdx.x * blockDim.x + threadIdx.x;
    if (e < ne) atomicAdd(&deg[(int)dst[e]], 1.0f);
}

// scatter-add h[src] rows into agg[dst]; one thread = (edge, 4 channels)
__global__ void k_scatter(const long long* __restrict__ src, const long long* __restrict__ dst,
                          const float* __restrict__ h, float* __restrict__ agg, int ne) {
    long long t = (long long)blockIdx.x * blockDim.x + threadIdx.x;
    int e = (int)(t >> 6);
    if (e >= ne) return;
    int c = ((int)t & 63) * 4;
    int s = (int)src[e], d = (int)dst[e];
    const float4 v = *(const float4*)(h + (long long)s * HID + c);
    float* a = agg + (long long)d * HID + c;
    atomicAdd(a + 0, v.x); atomicAdd(a + 1, v.y);
    atomicAdd(a + 2, v.z); atomicAdd(a + 3, v.w);
}

// agg / max(deg,1) -> bf16 row-major (WMMA A operand source)
__global__ void k_prep_agg(const float* __restrict__ agg, const float* __restrict__ deg,
                           unsigned short* __restrict__ aggbf, int n) {
    int i = blockIdx.x * blockDim.x + threadIdx.x;
    if (i >= n) return;
    float d = deg[i >> 8];
    d = d < 1.0f ? 1.0f : d;
    aggbf[i] = f2bf(agg[i] / d);
}

// input projection: h = x @ Wp + bp  (K=16, VALU is fine; also emit bf16 copy)
__global__ __launch_bounds__(256) void k_input_proj(
    const float* __restrict__ x, const float* __restrict__ Wp, const float* __restrict__ bp,
    float* __restrict__ h, unsigned short* __restrict__ hbf) {
    __shared__ float xs[IN_DIM];
    int n = blockIdx.x, c = threadIdx.x;
    if (c < IN_DIM) xs[c] = x[n * IN_DIM + c];
    __syncthreads();
    float acc = bp[c];
#pragma unroll
    for (int k = 0; k < IN_DIM; ++k) acc += xs[k] * Wp[k * HID + c];
    long long idx = (long long)n * HID + c;
    h[idx] = acc;
    hbf[idx] = f2bf(acc);
}

// Pack a 256x256 fp32 weight (row-major, [K][N]) into WMMA-B bf16 lane layout:
// element e = (((kt*NT + nt)*32) + lane)*16 + j
//   n = nt*16 + (lane&15);  k = kt*32 + (lane>>4)*16 + j   (contiguous k order, same as the A loads)
__global__ void k_pack_w(const float* __restrict__ W, unsigned short* __restrict__ P) {
    int e = blockIdx.x * blockDim.x + threadIdx.x;       // 65536 total
    int j    = e & 15;
    int lane = (e >> 4) & 31;
    int nt   = (e >> 9) & 15;
    int kt   = e >> 13;
    int nn = nt * 16 + (lane & 15);
    int kk = kt * 32 + (lane >> 4) * 16 + j;
    P[e] = f2bf(W[kk * HID + nn]);
}

// ------------------------------------------------------- WMMA GEMM kernel
// out[M,256] = A0 @ B0 (+ A1 @ B1 if DUAL), A row-major bf16, B pre-packed.
// One wave owns a 16-row tile; A (both operands, full K=256) preloaded in
// registers and reused across all 16 n-tiles. B tiles for each n-tile are
// staged cooperatively into LDS with ASYNC global->LDS loads, double
// buffered so the next stage's DMA overlaps the current stage's WMMAs.
template<int DUAL>
__global__ __launch_bounds__(256) void k_gemm_wmma(
    const unsigned short* __restrict__ A0, const unsigned short* __restrict__ A1,
    const unsigned short* __restrict__ B0, const unsigned short* __restrict__ B1,
    float* __restrict__ out) {
    // stage = one n-tile's B for all 8 k-tiles (and both operands if DUAL):
    //   per op: KT * 32 lanes * 16 bf16 = 4096 ushorts (8 KB)
    __shared__ unsigned short sB[2][(DUAL ? 2 : 1) * KT * 32 * 16];
    constexpr int OPS      = DUAL ? 2 : 1;
    constexpr int STAGE_US = OPS * KT * 32 * 16;            // ushorts per stage
    constexpr int NLOADS   = (STAGE_US * 2) / (256 * 16);   // b128 loads per thread

    unsigned short* lbase = &sB[0][0];
    int wave = threadIdx.x >> 5, lane = threadIdx.x & 31;
    int tile = blockIdx.x * 8 + wave;
    bool active = tile < ROW_TILES;
    if (!active) tile = ROW_TILES - 1;   // keep all threads for barriers/staging
    int m = lane & 15, half = lane >> 4;

    const unsigned short* ar0 = A0 + (long long)(tile * 16 + m) * HID;
    const unsigned short* ar1 = (DUAL ? A1 : A0) + (long long)(tile * 16 + m) * HID;

    // A layout (16-bit 16x32): lane half selects K sub-block; two 16B chunks.
    BF16x16 a0[KT], a1[KT];
#pragma unroll
    for (int kt = 0; kt < KT; ++kt) {
        int kb = kt * 32 + half * 16;
        a0[kt].q[0] = *(const U4*)(ar0 + kb);
        a0[kt].q[1] = *(const U4*)(ar0 + kb + 8);
        if (DUAL) {
            a1[kt].q[0] = *(const U4*)(ar1 + kb);
            a1[kt].q[1] = *(const U4*)(ar1 + kb + 8);
        }
    }

    // async-stage one n-tile's B into LDS buffer `buf`
    auto stage = [&](int nt, int buf) {
#pragma unroll
        for (int i = 0; i < NLOADS; ++i) {
            int L   = threadIdx.x + 256 * i;           // b128-chunk index
            int op  = DUAL ? (L >> 9) : 0;             // 512 chunks per op
            int rem = DUAL ? (L & 511) : L;
            int kt  = rem >> 6;                        // 64 chunks per k-tile
            int sub = rem & 63;
            const unsigned short* gsrc =
                (op ? B1 : B0) + ((kt * NT + nt) * 512) + sub * 8;
            unsigned short* ldst =
                lbase + (buf * STAGE_US + op * (KT * 512) + kt * 512 + sub * 8);
            __builtin_amdgcn_global_load_async_to_lds_b128(
                (v4i*)gsrc, (v4i*)ldst, 0, 0);
        }
    };

    stage(0, 0);
#pragma unroll 1
    for (int nt = 0; nt < NT; ++nt) {
        int buf = nt & 1;
        if (nt + 1 < NT) {
            stage(nt + 1, buf ^ 1);
            asm volatile("s_wait_asynccnt %0" :: "i"(NLOADS) : "memory");
        } else {
            asm volatile("s_wait_asynccnt 0" ::: "memory");
        }
        __syncthreads();                 // stage(nt) visible to all waves

        v8f acc = {};
#pragma unroll
        for (int kt = 0; kt < KT; ++kt) {
            BF16x16 b0;
            const unsigned short* bp0 = &sB[buf][kt * 512 + lane * 16];
            b0.q[0] = *(const U4*)(bp0);
            b0.q[1] = *(const U4*)(bp0 + 8);
            acc = __builtin_amdgcn_wmma_f32_16x16x32_bf16(
                false, a0[kt].v, false, b0.v, (short)0, acc, false, false);
            if (DUAL) {
                BF16x16 b1;
                const unsigned short* bp1 = &sB[buf][KT * 512 + kt * 512 + lane * 16];
                b1.q[0] = *(const U4*)(bp1);
                b1.q[1] = *(const U4*)(bp1 + 8);
                acc = __builtin_amdgcn_wmma_f32_16x16x32_bf16(
                    false, a1[kt].v, false, b1.v, (short)0, acc, false, false);
            }
        }
        __syncthreads();                 // protect buf reuse by stage(nt+2)

        if (active) {
            // C/D layout: VGPR r -> rows (half*8 + r), cols nt*16 + m
            int row = tile * 16 + half * 8;
            int col = nt * 16 + m;
#pragma unroll
            for (int r = 0; r < 8; ++r)
                out[(long long)(row + r) * HID + col] = acc[r];
        }
    }
}

// ---------------------------------------------- per-row epilogue (fp32)
// v = out + bl ; u = v / max(||v||2, 1e-12) ; t = LN(u)*g + b ; h = silu(t) + h_in
__global__ __launch_bounds__(256) void k_epilogue(
    const float* __restrict__ outb, const float* __restrict__ bl,
    const float* __restrict__ g, const float* __restrict__ b,
    float* __restrict__ h, unsigned short* __restrict__ hbf) {
    __shared__ float red[HID];
    int n = blockIdx.x, c = threadIdx.x;
    long long idx = (long long)n * HID + c;
    float v = outb[idx] + bl[c];

    red[c] = v * v; __syncthreads();
    for (int s = 128; s > 0; s >>= 1) { if (c < s) red[c] += red[c + s]; __syncthreads(); }
    float nrm = sqrtf(red[0]); __syncthreads();
    float u = v / fmaxf(nrm, 1e-12f);

    red[c] = u; __syncthreads();
    for (int s = 128; s > 0; s >>= 1) { if (c < s) red[c] += red[c + s]; __syncthreads(); }
    float mu = red[0] * (1.0f / HID); __syncthreads();

    float d = u - mu;
    red[c] = d * d; __syncthreads();
    for (int s = 128; s > 0; s >>= 1) { if (c < s) red[c] += red[c + s]; __syncthreads(); }
    float var = red[0] * (1.0f / HID);

    float t = d * rsqrtf(var + 1e-5f) * g[c] + b[c];
    float si = t / (1.0f + expf(-t));          // silu
    float hn = si + h[idx];                    // residual
    h[idx] = hn;
    hbf[idx] = f2bf(hn);
}

// head tail: out[n,0:3] = silu(o1 + b1) @ W2 + b2
__global__ __launch_bounds__(256) void k_head(
    const float* __restrict__ o1, const float* __restrict__ b1,
    const float* __restrict__ W2, const float* __restrict__ b2,
    float* __restrict__ out) {
    __shared__ float red[3][HID];
    int n = blockIdx.x, c = threadIdx.x;
    float t = o1[(long long)n * HID + c] + b1[c];
    float s = t / (1.0f + expf(-t));
    red[0][c] = s * W2[c * 3 + 0];
    red[1][c] = s * W2[c * 3 + 1];
    red[2][c] = s * W2[c * 3 + 2];
    __syncthreads();
    for (int st = 128; st > 0; st >>= 1) {
        if (c < st) {
            red[0][c] += red[0][c + st];
            red[1][c] += red[1][c + st];
            red[2][c] += red[2][c + st];
        }
        __syncthreads();
    }
    if (c < 3) out[n * 3 + c] = red[c][0] + b2[c];
}

// ---------------------------------------------------------------- driver
extern "C" void kernel_launch(void* const* d_in, const int* in_sizes, int n_in,
                              void* d_out, int out_size, void* d_ws, size_t ws_size,
                              hipStream_t stream) {
    (void)in_sizes; (void)n_in; (void)out_size; (void)ws_size;
    const float*     x    = (const float*)d_in[0];
    const long long* ei   = (const long long*)d_in[1];   // int64 [2, N_EDGES]
    const float*     Wp   = (const float*)d_in[2];
    const float*     bp   = (const float*)d_in[3];
    const float*     Wl   = (const float*)d_in[4];
    const float*     bl   = (const float*)d_in[5];
    const float*     Wr   = (const float*)d_in[6];
    const float*     ln_g = (const float*)d_in[7];
    const float*     ln_b = (const float*)d_in[8];
    const float*     W1   = (const float*)d_in[9];
    const float*     b1   = (const float*)d_in[10];
    const float*     W2   = (const float*)d_in[11];
    const float*     b2   = (const float*)d_in[12];
    const long long* src = ei;
    const long long* dst = ei + N_EDGES;

    // workspace carve-out (~155 MB)
    char* ws = (char*)d_ws;
    size_t off = 0;
    auto carve = [&](size_t bytes) -> char* {
        char* p = ws + off;
        off += (bytes + 255) & ~(size_t)255;
        return p;
    };
    float*          h     = (float*)carve((size_t)N_NODES * HID * 4);
    unsigned short* hbf   = (unsigned short*)carve((size_t)N_NODES * HID * 2);
    unsigned short* aggbf = (unsigned short*)carve((size_t)N_NODES * HID * 2);
    float*          agg   = (float*)carve((size_t)N_NODES * HID * 4);  // reused as GEMM out
    float*          outb  = agg;                                       // alias (agg dead after prep)
    float*          deg   = (float*)carve((size_t)N_NODES * 4);
    unsigned short* WlP   = (unsigned short*)carve((size_t)N_LAYERS * HID * HID * 2);
    unsigned short* WrP   = (unsigned short*)carve((size_t)N_LAYERS * HID * HID * 2);
    unsigned short* W1P   = (unsigned short*)carve((size_t)HID * HID * 2);

    const int elemHN = N_NODES * HID;           // 12.8M
    const int gemmBlocks = (ROW_TILES + 7) / 8; // 8 waves/block, 1 row-tile/wave

    // degree (once)
    k_zero<<<(N_NODES + 255) / 256, 256, 0, stream>>>(deg, N_NODES);
    k_degree<<<(N_EDGES + 255) / 256, 256, 0, stream>>>(dst, deg, N_EDGES);

    // pre-pack weights to WMMA-B bf16 layout (once per launch)
    for (int i = 0; i < N_LAYERS; ++i) {
        k_pack_w<<<256, 256, 0, stream>>>(Wl + (size_t)i * HID * HID, WlP + (size_t)i * HID * HID);
        k_pack_w<<<256, 256, 0, stream>>>(Wr + (size_t)i * HID * HID, WrP + (size_t)i * HID * HID);
    }
    k_pack_w<<<256, 256, 0, stream>>>(W1, W1P);

    // input projection
    k_input_proj<<<N_NODES, 256, 0, stream>>>(x, Wp, bp, h, hbf);

    // SAGE layers
    for (int i = 0; i < N_LAYERS; ++i) {
        k_zero<<<(elemHN + 255) / 256, 256, 0, stream>>>(agg, elemHN);
        k_scatter<<<(int)(((long long)N_EDGES * 64 + 255) / 256), 256, 0, stream>>>(src, dst, h, agg, N_EDGES);
        k_prep_agg<<<(elemHN + 255) / 256, 256, 0, stream>>>(agg, deg, aggbf, elemHN);
        k_gemm_wmma<1><<<gemmBlocks, 256, 0, stream>>>(
            aggbf, hbf, WlP + (size_t)i * HID * HID, WrP + (size_t)i * HID * HID, outb);
        k_epilogue<<<N_NODES, 256, 0, stream>>>(
            outb, bl + (size_t)i * HID, ln_g + (size_t)i * HID, ln_b + (size_t)i * HID, h, hbf);
    }

    // output head
    k_gemm_wmma<0><<<gemmBlocks, 256, 0, stream>>>(hbf, nullptr, W1P, nullptr, outb);
    k_head<<<N_NODES, 256, 0, stream>>>(outb, b1, W2, b2, (float*)d_out);
}
